// MultiHeadAttentionFused_678604833528
// MI455X (gfx1250) — matrix operand order repaired and checked by reading, not compiled, not run on hardware
//
#include <hip/hip_runtime.h>
#include <hip/hip_bf16.h>

// ---------------------------------------------------------------------------
// MI455X fused MHA: bf16 WMMA GEMMs (async-LDS double-buffered, peeled
// software pipeline) + fp32-accumulate flash attention.
// All bf16 buffers are passed as unsigned short* (host never touches __bf16).
// ---------------------------------------------------------------------------

typedef __attribute__((ext_vector_type(16))) __bf16 v16bf;
typedef __attribute__((ext_vector_type(8)))  __bf16 v8bf;
typedef __attribute__((ext_vector_type(8)))  float  v8f;

#define D_MODEL   2048
#define NUM_HEADS 16
#define HEAD_DIM  128
#define BATCH     2
#define SEQ_L     2048
#define BL        (BATCH * SEQ_L)     // 4096 tokens

union V16BF {
    v8bf     h[2];
    v16bf    v;
    unsigned u[8];
};

__device__ __forceinline__ v8f zero8() {
    v8f z;
#pragma unroll
    for (int i = 0; i < 8; ++i) z[i] = 0.0f;
    return z;
}

// Async 16-byte global -> LDS copy (CDNA5 GLOBAL_LOAD_ASYNC_TO_LDS_B128,
// tracked by ASYNCcnt).  Inline asm per bridge-doc guidance (portable across
// toolchains, bypasses builtin arity differences).
__device__ __forceinline__ void async_copy16(const __bf16* g, const __bf16* l) {
    const unsigned            lds_off = (unsigned)(uintptr_t)l;   // low 32 = LDS offset
    const unsigned long long  gaddr   = (unsigned long long)(uintptr_t)g;
    asm volatile("global_load_async_to_lds_b128 %0, %1, off"
                 :: "v"(lds_off), "v"(gaddr)
                 : "memory");
}

// ---------------------------------------------------------------------------
// fp32 -> bf16 cast (grid-stride)
// ---------------------------------------------------------------------------
__global__ void k_cast_bf16(const float* __restrict__ src,
                            unsigned short* __restrict__ dst, int n) {
    __bf16* d = (__bf16*)dst;
    int i = blockIdx.x * blockDim.x + threadIdx.x;
    int stride = gridDim.x * blockDim.x;
    for (; i < n; i += stride) d[i] = (__bf16)src[i];
}

// ---------------------------------------------------------------------------
// C[m,n] = sum_k A[m,k] * W[n,k] + bias[n]     (A, W bf16; C fp32; K = 2048)
// Block tile 128x128, 8 waves (4 along M x 2 along N), wave tile 32x64.
// A/B 128x32 tiles staged in LDS via async b128 copies, double buffered;
// peeled software pipeline: branch-free steady state (unroll 1 so the WMMA
// accumulators stay as in-place loop-carried registers), prefetch via running
// pointers, fenced with s_wait_asynccnt + workgroup split barrier.
// Fragment layouts per CDNA5 ISA 7.12.2 (16-bit A 16x32, B 32x16, f32 C/D).
// ---------------------------------------------------------------------------
__global__ __launch_bounds__(256)
void k_gemm_bf16(const unsigned short* __restrict__ A16,
                 const unsigned short* __restrict__ W16,
                 const float* __restrict__ bias,
                 float* __restrict__ C,
                 int N) {
    constexpr int K     = 2048;
    constexpr int NSTEP = K / 32;            // 64
    constexpr int BUFE  = 128 * 32;          // elements per LDS buffer

    const __bf16* A = (const __bf16*)A16;
    const __bf16* W = (const __bf16*)W16;

    __shared__ __bf16 Asm[2][BUFE];          // 8 KB per buffer
    __shared__ __bf16 Bsm[2][BUFE];          // 8 KB per buffer

    const int tid   = threadIdx.x;
    const int lane  = tid & 31;
    const int wave  = tid >> 5;
    const int waveM = wave & 3;              // 4 waves along M
    const int waveN = wave >> 2;             // 2 waves along N
    const int mTile = blockIdx.y * 128;
    const int nTile = blockIdx.x * 128;
    const int l15   = lane & 15;
    const int hi    = lane >> 4;

    // copy work decomposition: 512 16B chunks per tile, 2 per thread
    const int rC = tid >> 2;                 // row 0..63 (and +64)
    const int oC = (tid & 3) * 8;            // element offset within 32-K row

    // running global prefetch pointers (advance 32 elements per K-step)
    const __bf16* gA0 = A + (size_t)(mTile + rC) * K + oC;
    const __bf16* gA1 = gA0 + (size_t)64 * K;
    const __bf16* gB0 = W + (size_t)(nTile + rC) * K + oC;
    const __bf16* gB1 = gB0 + (size_t)64 * K;

    // fixed LDS destinations (element offsets within buffer 0)
    __bf16* lA = &Asm[0][rC * 32 + oC];
    __bf16* lB = &Bsm[0][rC * 32 + oC];

    auto issue_tiles = [&](int nb) {
        const int bo = nb * BUFE;
        async_copy16(gA0, lA + bo);
        async_copy16(gA1, lA + bo + 64 * 32);
        async_copy16(gB0, lB + bo);
        async_copy16(gB1, lB + bo + 64 * 32);
        gA0 += 32; gA1 += 32; gB0 += 32; gB1 += 32;
    };

    v8f acc[2][4];
#pragma unroll
    for (int i = 0; i < 2; ++i)
#pragma unroll
        for (int j = 0; j < 4; ++j) acc[i][j] = zero8();

    // per-wave LDS read offsets (element units, constant across the K loop)
    const int aOff = (waveM * 32 + l15) * 32 + (hi ? 8 : 0);
    const int bOff = (waveN * 64 + l15) * 32 + (hi ? 8 : 0);

    auto compute = [&](int buf) {
        const __bf16* Abase = &Asm[buf][aOff];
        const __bf16* Bbase = &Bsm[buf][bOff];
        // A-fragments: lanes 0-15 K {0..7,16..23}, lanes 16-31 K {8..15,24..31}
        V16BF a[2];
#pragma unroll
        for (int mi = 0; mi < 2; ++mi) {
            const __bf16* p = Abase + mi * (16 * 32);
            a[mi].h[0] = *(const v8bf*)p;
            a[mi].h[1] = *(const v8bf*)(p + 16);
        }
#pragma unroll
        for (int nj = 0; nj < 4; ++nj) {
            // B-fragment: lane = column, lanes 0-15 K {0..7,16..23}, lanes 16-31 K {8..15,24..31}
            const __bf16* p = Bbase + nj * (16 * 32);
            V16BF b;
            b.h[0] = *(const v8bf*)p;
            b.h[1] = *(const v8bf*)(p + 16);
            acc[0][nj] = __builtin_amdgcn_wmma_f32_16x16x32_bf16(
                false, a[0].v, false, b.v, (short)0, acc[0][nj], false, false);
            acc[1][nj] = __builtin_amdgcn_wmma_f32_16x16x32_bf16(
                false, a[1].v, false, b.v, (short)0, acc[1][nj], false, false);
        }
    };

    // ---- peeled pipeline: prologue fill, branch-free steady state, epilogue
    issue_tiles(0);
#pragma unroll 1
    for (int s = 0; s < NSTEP - 1; ++s) {
        issue_tiles((s + 1) & 1);            // prefetch next tile
        // retire the 4 older async copies (current buffer), keep 4 in flight
        asm volatile("s_wait_asynccnt 0x4" ::: "memory");
        __syncthreads();                     // all waves' tile data visible
        compute(s & 1);
        __syncthreads();                     // buffer free for next prefetch
    }
    asm volatile("s_wait_asynccnt 0x0" ::: "memory");
    __syncthreads();
    compute((NSTEP - 1) & 1);

    // C/D layout: VGPR r -> row r (lanes 0-15) or r+8 (lanes 16-31); col = lane%16
    const int mBase = mTile + waveM * 32;
    const int nBase = nTile + waveN * 64;
#pragma unroll
    for (int mi = 0; mi < 2; ++mi) {
#pragma unroll
        for (int nj = 0; nj < 4; ++nj) {
            const int col = nBase + nj * 16 + l15;
            const float bv = bias[col];
#pragma unroll
            for (int r = 0; r < 8; ++r) {
                const int row = mBase + mi * 16 + r + hi * 8;
                C[(size_t)row * N + col] = acc[mi][nj][r] + bv;
            }
        }
    }
}

// ---------------------------------------------------------------------------
// RoPE (rotate-half) on Q and K, writing head-major bf16 [b,h,l,d].
// One thread per (b,l,h,pair j in [0,64)).  qkv layout: [b,l, 3*2048].
// ---------------------------------------------------------------------------
__global__ void k_rope(const float* __restrict__ qkv,
                       unsigned short* __restrict__ Qb16,
                       unsigned short* __restrict__ Kb16) {
    __bf16* Qb = (__bf16*)Qb16;
    __bf16* Kb = (__bf16*)Kb16;
    const int idx = blockIdx.x * blockDim.x + threadIdx.x;   // 2^22 threads
    const int j = idx & 63;
    const int h = (idx >> 6) & 15;
    const int l = (idx >> 10) & 2047;
    const int b = idx >> 21;

    const float* base =
        qkv + ((size_t)(b * SEQ_L + l)) * (3 * D_MODEL) + h * HEAD_DIM;

    // inv_freq = 10000^(-(2j)/128)
    const float inv = __expf(-(float)(2 * j) * (1.0f / 128.0f) * 9.21034037198f);
    const float theta = (float)l * inv;
    float s, c;
    __sincosf(theta, &s, &c);

    const float q1 = base[j],           q2 = base[j + 64];
    const float k1 = base[D_MODEL + j], k2 = base[D_MODEL + j + 64];

    __bf16* Qo = Qb + (((size_t)(b * NUM_HEADS + h)) * SEQ_L + l) * HEAD_DIM;
    __bf16* Ko = Kb + (((size_t)(b * NUM_HEADS + h)) * SEQ_L + l) * HEAD_DIM;
    Qo[j]      = (__bf16)(q1 * c - q2 * s);
    Qo[j + 64] = (__bf16)(q2 * c + q1 * s);
    Ko[j]      = (__bf16)(k1 * c - k2 * s);
    Ko[j + 64] = (__bf16)(k2 * c + k1 * s);
}

// ---------------------------------------------------------------------------
// V transpose to [b,h,d,l] bf16 so P*V B-fragments load contiguous in key dim.
// ---------------------------------------------------------------------------
__global__ void k_transpose_v(const float* __restrict__ qkv,
                              unsigned short* __restrict__ Vt16) {
    __bf16* Vt = (__bf16*)Vt16;
    const int idx = blockIdx.x * blockDim.x + threadIdx.x;   // 2^23 threads
    const int l = idx & 2047;
    const int d = (idx >> 11) & 127;
    const int h = (idx >> 18) & 15;
    const int b = idx >> 22;
    Vt[(((size_t)(b * NUM_HEADS + h)) * HEAD_DIM + d) * SEQ_L + l] =
        (__bf16)qkv[((size_t)(b * SEQ_L + l)) * (3 * D_MODEL) +
                    2 * D_MODEL + h * HEAD_DIM + d];
}

// ---------------------------------------------------------------------------
// Flash attention: each wave owns a 16-row query tile of one (b,h), streams
// causal key blocks of 32.  S and P*V through v_wmma_f32_16x16x32_bf16; the
// D-layout -> A-layout conversion of P goes through LDS (per-wave 1 KB tile).
// ---------------------------------------------------------------------------
__global__ __launch_bounds__(128)
void k_attn(const unsigned short* __restrict__ Qb16,
            const unsigned short* __restrict__ Kb16,
            const unsigned short* __restrict__ Vt16,
            unsigned short* __restrict__ Ctx16) {
    const __bf16* Qb = (const __bf16*)Qb16;
    const __bf16* Kb = (const __bf16*)Kb16;
    const __bf16* Vt = (const __bf16*)Vt16;
    __bf16* Ctx = (__bf16*)Ctx16;

    __shared__ __bf16 Plds[4][16 * 32];      // one 16x32 bf16 tile per wave

    const int lane = threadIdx.x & 31;
    const int wv   = threadIdx.x >> 5;
    const int gw   = blockIdx.x * 4 + wv;
    const int nQT  = SEQ_L / 16;             // 128 query tiles per (b,h)
    const int qt   = gw % nQT;
    const int h    = (gw / nQT) % NUM_HEADS;
    const int b    = gw / (nQT * NUM_HEADS);
    const int l15  = lane & 15;
    const int hi   = lane >> 4;

    const __bf16* Qbase = Qb + ((size_t)(b * NUM_HEADS + h)) * SEQ_L * HEAD_DIM;
    const __bf16* Kbase = Kb + ((size_t)(b * NUM_HEADS + h)) * SEQ_L * HEAD_DIM;
    const __bf16* Vbase = Vt + ((size_t)(b * NUM_HEADS + h)) * HEAD_DIM * SEQ_L;

    // Load the 16x128 Q tile as 4 A-fragments (kept in VGPRs for whole pass)
    V16BF qf[4];
    {
        const __bf16* Ar = Qbase + (size_t)(qt * 16 + l15) * HEAD_DIM + (hi ? 8 : 0);
#pragma unroll
        for (int kd = 0; kd < 4; ++kd) {
            qf[kd].h[0] = *(const v8bf*)(Ar + kd * 32);
            qf[kd].h[1] = *(const v8bf*)(Ar + kd * 32 + 16);
        }
    }

    v8f o[8];
#pragma unroll
    for (int f = 0; f < 8; ++f) o[f] = zero8();
    float mrow[8], lrow[8];
#pragma unroll
    for (int r = 0; r < 8; ++r) { mrow[r] = -3.0e38f; lrow[r] = 0.0f; }

    // softmax(S/sqrt(128)) via exp2: cexp = scale * log2(e)
    const float cexp = 0.08838834764831845f * 1.4426950408889634f;

    const int lastKB = (qt * 16 + 15) >> 5;
    for (int kb = 0; kb <= lastKB; ++kb) {
        // ---- S = Q K^T (16x32 tile, two 16x16 fragments) ----
        v8f s[2];
        s[0] = zero8(); s[1] = zero8();
#pragma unroll
        for (int n = 0; n < 2; ++n) {
            const __bf16* Kp = Kbase +
                (size_t)(kb * 32 + n * 16 + l15) * HEAD_DIM + (hi ? 8 : 0);
#pragma unroll
            for (int kd = 0; kd < 4; ++kd) {
                V16BF bk;
                bk.h[0] = *(const v8bf*)(Kp + kd * 32);
                bk.h[1] = *(const v8bf*)(Kp + kd * 32 + 16);
                s[n] = __builtin_amdgcn_wmma_f32_16x16x32_bf16(
                    false, qf[kd].v, false, bk.v, (short)0, s[n], false, false);
            }
        }

        // ---- causal mask + online softmax (row = VGPR idx, col = lane) ----
#pragma unroll
        for (int r = 0; r < 8; ++r) {
            const int qr   = qt * 16 + r + hi * 8;
            const int key0 = kb * 32 + l15;
            const int key1 = key0 + 16;
            const float s0 = (key0 <= qr) ? s[0][r] : -3.0e38f;
            const float s1 = (key1 <= qr) ? s[1][r] : -3.0e38f;

            float v = fmaxf(s0, s1);
            v = fmaxf(v, __shfl_xor(v, 1, 32));
            v = fmaxf(v, __shfl_xor(v, 2, 32));
            v = fmaxf(v, __shfl_xor(v, 4, 32));
            v = fmaxf(v, __shfl_xor(v, 8, 32));

            const float mnew  = fmaxf(mrow[r], v);
            const float alpha = exp2f((mrow[r] - mnew) * cexp);
            const float p0 = exp2f((s0 - mnew) * cexp);
            const float p1 = exp2f((s1 - mnew) * cexp);

            float ps = p0 + p1;
            ps += __shfl_xor(ps, 1, 32);
            ps += __shfl_xor(ps, 2, 32);
            ps += __shfl_xor(ps, 4, 32);
            ps += __shfl_xor(ps, 8, 32);

            lrow[r] = lrow[r] * alpha + ps;
            mrow[r] = mnew;
#pragma unroll
            for (int f = 0; f < 8; ++f) o[f][r] *= alpha;

            // stage P (bf16) into the per-wave LDS tile, row-major [16][32]
            const int m = r + hi * 8;
            Plds[wv][m * 32 + l15]      = (__bf16)p0;
            Plds[wv][m * 32 + 16 + l15] = (__bf16)p1;
        }

        // ---- reload P as a 16x32 A-fragment (pairs of K are one b32) ----
        V16BF pa;
        {
            const unsigned* pl = (const unsigned*)&Plds[wv][0];
            const int m  = l15;
            const int b0 = hi ? 4 : 0;          // K base / 2
#pragma unroll
            for (int t = 0; t < 4; ++t) {
                pa.u[t]     = pl[m * 16 + b0 + t];          // K 0..7  (+8 if hi)
                pa.u[4 + t] = pl[m * 16 + 8 + b0 + t];      // K 16..23 (+8 if hi)
            }
        }

        // ---- O += P V  (8 head-dim column blocks of 16) ----
#pragma unroll
        for (int nd = 0; nd < 8; ++nd) {
            const __bf16* Vp = Vbase +
                (size_t)(nd * 16 + l15) * SEQ_L + kb * 32 + (hi ? 8 : 0);
            V16BF bv;
            bv.h[0] = *(const v8bf*)(Vp);
            bv.h[1] = *(const v8bf*)(Vp + 16);
            o[nd] = __builtin_amdgcn_wmma_f32_16x16x32_bf16(
                false, pa.v, false, bv.v, (short)0, o[nd], false, false);
        }
    }

    // ---- epilogue: normalize and scatter back to token-major [b,l,2048] ----
#pragma unroll
    for (int r = 0; r < 8; ++r) {
        const float inv = 1.0f / lrow[r];
        const int qr = qt * 16 + r + hi * 8;
        __bf16* Cp = Ctx + ((size_t)(b * SEQ_L + qr)) * D_MODEL + h * HEAD_DIM + l15;
#pragma unroll
        for (int f = 0; f < 8; ++f) Cp[f * 16] = (__bf16)(o[f][r] * inv);
    }
}

// ---------------------------------------------------------------------------
// Launcher
// ---------------------------------------------------------------------------
extern "C" void kernel_launch(void* const* d_in, const int* in_sizes, int n_in,
                              void* d_out, int out_size, void* d_ws, size_t ws_size,
                              hipStream_t stream) {
    const float* query = (const float*)d_in[0];   // [2,2048,2048]
    const float* W_qkv = (const float*)d_in[1];   // [6144,2048]
    const float* b_qkv = (const float*)d_in[2];   // [6144]
    const float* W_out = (const float*)d_in[3];   // [2048,2048]
    const float* b_out = (const float*)d_in[4];   // [2048]
    float* out = (float*)d_out;                   // [2,2048,2048] fp32

    // Workspace carve-up (all sizes are multiples of 2 MB -> aligned)
    char* ws = (char*)d_ws;
    unsigned short* q16    = (unsigned short*)ws; ws += (size_t)BL * D_MODEL * 2;
    unsigned short* Wqkv16 = (unsigned short*)ws; ws += (size_t)3 * D_MODEL * D_MODEL * 2;
    unsigned short* Wout16 = (unsigned short*)ws; ws += (size_t)D_MODEL * D_MODEL * 2;
    float*          qkv    = (float*)ws;          ws += (size_t)BL * 3 * D_MODEL * 4;
    unsigned short* Qb     = (unsigned short*)ws; ws += (size_t)BL * D_MODEL * 2;
    unsigned short* Kb     = (unsigned short*)ws; ws += (size_t)BL * D_MODEL * 2;
    unsigned short* Vt     = (unsigned short*)ws; ws += (size_t)BL * D_MODEL * 2;
    unsigned short* attn16 = q16;                 // reuse: q16 dead after QKV GEMM

    // 1) casts to bf16
    k_cast_bf16<<<4096, 256, 0, stream>>>(query, q16, BL * D_MODEL);
    k_cast_bf16<<<4096, 256, 0, stream>>>(W_qkv, Wqkv16, 3 * D_MODEL * D_MODEL);
    k_cast_bf16<<<4096, 256, 0, stream>>>(W_out, Wout16, D_MODEL * D_MODEL);

    // 2) QKV GEMM: [4096 x 2048] x [6144 x 2048]^T -> fp32 [4096 x 6144]
    {
        dim3 grid((3 * D_MODEL) / 128, BL / 128);
        k_gemm_bf16<<<grid, 256, 0, stream>>>(q16, Wqkv16, b_qkv, qkv,
                                              3 * D_MODEL);
    }

    // 3) RoPE + head-major packing; V transpose
    k_rope<<<(1 << 22) / 256, 256, 0, stream>>>(qkv, Qb, Kb);
    k_transpose_v<<<(1 << 23) / 256, 256, 0, stream>>>(qkv, Vt);

    // 4) causal flash attention -> bf16 context [4096 x 2048]
    k_attn<<<(BATCH * NUM_HEADS * (SEQ_L / 16)) / 4, 128, 0, stream>>>(
        Qb, Kb, Vt, attn16);

    // 5) output projection: [4096 x 2048] x [2048 x 2048]^T + b_out -> d_out
    {
        dim3 grid(D_MODEL / 128, BL / 128);
        k_gemm_bf16<<<grid, 256, 0, stream>>>(attn16, Wout16, b_out, out,
                                              D_MODEL);
    }
}
